// GraphConvBlock_78752520339638
// MI455X (gfx1250) — matrix operand order repaired and checked by reading, not compiled, not run on hardware
//
#include <hip/hip_runtime.h>
#include <math.h>

#define NNODES 50000
#define NEDGES 500000
#define DMODEL 128
#define DINNER 256
#define NEXP   64
#define PAIRS  (NNODES * 2)
#define CAP    3126            /* 2 * ceil(100000/64) */
#define NT     ((CAP + 15) / 16)
#define BNEPS  1e-5f

typedef __attribute__((ext_vector_type(16))) __bf16         v16bf;
typedef __attribute__((ext_vector_type(8)))  float          v8f;
typedef __attribute__((ext_vector_type(8)))  unsigned short u16x8;

__device__ __forceinline__ unsigned short f2bf(float f) {
    union { float f; unsigned u; } v; v.f = f;
    unsigned u = v.u;
    u += 0x7FFFu + ((u >> 16) & 1u);          // round-to-nearest-even
    return (unsigned short)(u >> 16);
}
__device__ __forceinline__ float bf2f(unsigned short h) {
    union { unsigned u; float f; } v; v.u = ((unsigned)h) << 16;
    return v.f;
}

// ---- WMMA fragment loaders (ISA 7.12.2 16-bit layouts) ----
// A tile in LDS row-major [16][lda] ushort. Lane l<16: K=0..7 & 16..23; l>=16: +8.
__device__ __forceinline__ v16bf frag_a(const unsigned short* A, int lda, int lane, int k0) {
    int ho = (lane >> 4) * 8, m = lane & 15;
    const unsigned short* p = A + m * lda + k0 + ho;
    union { v16bf v; u16x8 h[2]; } u;
    u.h[0] = *(const u16x8*)(p);
    u.h[1] = *(const u16x8*)(p + 16);
    return u.v;
}
// B staged TRANSPOSED in LDS [n][ldt] ushort. Lane l<16: K=0..7 & 16..23; l>=16: +8.
__device__ __forceinline__ v16bf frag_bt(const unsigned short* Bt, int ldt, int lane, int k0, int n0) {
    int ho = (lane >> 4) * 8, n = lane & 15;
    const unsigned short* p = Bt + (n0 + n) * ldt + k0 + ho;
    union { v16bf v; u16x8 h[2]; } u;
    u.h[0] = *(const u16x8*)(p);
    u.h[1] = *(const u16x8*)(p + 16);
    return u.v;
}
__device__ __forceinline__ v8f wmma_bf16(v16bf a, v16bf b, v8f c) {
    return __builtin_amdgcn_wmma_f32_16x16x32_bf16(false, a, false, b, (short)0, c, false, false);
}

// ---------------- GCN aggregation ----------------
__global__ void k_degree(const int* __restrict__ dst, float* __restrict__ deg, int n) {
    for (int i = blockIdx.x * blockDim.x + threadIdx.x; i < n; i += gridDim.x * blockDim.x)
        __hip_atomic_fetch_add(&deg[dst[i]], 1.0f, __ATOMIC_RELAXED, __HIP_MEMORY_SCOPE_AGENT);
}

__global__ __launch_bounds__(128) void k_scatter(const float* __restrict__ feats,
                                                 const int* __restrict__ src,
                                                 const int* __restrict__ dst,
                                                 const float* __restrict__ deg,
                                                 float* __restrict__ agg) {
    __shared__ float snorm; __shared__ int ssrc, sdst;
    int e = blockIdx.x;
    if (threadIdx.x == 0) {
        int s = src[e], d = dst[e];
        ssrc = s; sdst = d;
        float ds = deg[s] < 1.f ? 1.f : deg[s];
        float dd = deg[d] < 1.f ? 1.f : deg[d];
        snorm = rsqrtf(ds) * rsqrtf(dd);
    }
    __syncthreads();
    float v = feats[(size_t)ssrc * DMODEL + threadIdx.x] * snorm;
    __hip_atomic_fetch_add(&agg[(size_t)sdst * DMODEL + threadIdx.x], v,
                           __ATOMIC_RELAXED, __HIP_MEMORY_SCOPE_AGENT);
}

// ---------------- conv GEMM: nf = agg @ Wconv + b (50000x128 * 128x128) ----------------
__global__ __launch_bounds__(256) void k_convgemm(const float* __restrict__ A,
                                                  const float* __restrict__ W,
                                                  const float* __restrict__ bias,
                                                  float* __restrict__ out) {
    __shared__ unsigned short sA[16 * 136];
    __shared__ unsigned short sBt[128 * 136];   // transposed: [n][k]
    int tid = threadIdx.x;
    int row0 = blockIdx.x * 16;
    for (int i = tid; i < 16 * 128; i += 256) {
        int r = i >> 7, c = i & 127;
        sA[r * 136 + c] = f2bf(A[((size_t)row0 + r) * DMODEL + c]);
    }
    for (int i = tid; i < 128 * 128; i += 256) {
        int k = i >> 7, n = i & 127;
        sBt[n * 136 + k] = f2bf(W[(size_t)k * DMODEL + n]);
    }
    __syncthreads();
    int lane = tid & 31, wave = tid >> 5;
    int half = lane >> 4, ln = lane & 15;
    int n0 = wave * 16;
    v8f acc = {};
#pragma unroll
    for (int k0 = 0; k0 < 128; k0 += 32)
        acc = wmma_bf16(frag_a(sA, 136, lane, k0), frag_bt(sBt, 136, lane, k0, n0), acc);
    int col = n0 + ln;
#pragma unroll
    for (int r = 0; r < 8; ++r) {
        int m = r + 8 * half;
        out[((size_t)row0 + m) * DMODEL + col] = acc[r] + bias[col];
    }
}

// ---------------- gate + top2 + softmax ----------------
__global__ __launch_bounds__(256) void k_gate(const float* __restrict__ x,
                                              const float* __restrict__ gW,
                                              const float* __restrict__ gb,
                                              int* __restrict__ topi,
                                              float* __restrict__ gwout) {
    __shared__ float xs[4][128];
    __shared__ float sc[4][64];
    int tid = threadIdx.x;
    int g = tid >> 6, e = tid & 63;
    int t = blockIdx.x * 4 + g;
    xs[g][e]      = x[(size_t)t * DMODEL + e];
    xs[g][64 + e] = x[(size_t)t * DMODEL + 64 + e];
    __syncthreads();
    float acc = gb[e];
#pragma unroll 8
    for (int k = 0; k < 128; ++k) acc += xs[g][k] * gW[k * NEXP + e];
    sc[g][e] = acc;
    __syncthreads();
    if (e == 0) {
        float v1 = -1e30f, v2 = -1e30f; int i1 = 0, i2 = 0;
        for (int j = 0; j < 64; ++j) {
            float v = sc[g][j];
            if (v > v1) { v2 = v1; i2 = i1; v1 = v; i1 = j; }
            else if (v > v2) { v2 = v; i2 = j; }
        }
        float e1 = expf(v2 - v1);
        float d = 1.f + e1;
        topi[t * 2]     = i1;  topi[t * 2 + 1] = i2;
        gwout[t * 2]    = 1.f / d;  gwout[t * 2 + 1] = e1 / d;
    }
}

// ---------------- dispatch: rank per expert, gather rows to bf16 buffer ----------------
__global__ __launch_bounds__(128) void k_dispatch(const float* __restrict__ x,
                                                  const int* __restrict__ topi,
                                                  unsigned short* __restrict__ buf,
                                                  int* __restrict__ slot_map,
                                                  int* __restrict__ cnt) {
    int p = blockIdx.x, tid = threadIdx.x;
    __shared__ int sslot;
    if (tid == 0) {
        int e = topi[p];
        int pos = atomicAdd(&cnt[e], 1);
        int sl = (pos < CAP) ? e * CAP + pos : -1;
        sslot = sl;
        slot_map[p] = sl;
    }
    __syncthreads();
    int sl = sslot;
    if (sl >= 0) {
        int t = p >> 1;
        buf[(size_t)sl * DMODEL + tid] = f2bf(x[(size_t)t * DMODEL + tid]);
    }
}

// ---------------- fused expert FFN: gelu(X@W1+b1)@W2+b2 ----------------
__global__ __launch_bounds__(256) void k_expert(const unsigned short* __restrict__ buf,
                                                const float* __restrict__ W1,
                                                const float* __restrict__ b1,
                                                const float* __restrict__ W2,
                                                const float* __restrict__ b2,
                                                const int* __restrict__ cnt,
                                                unsigned short* __restrict__ ybuf) {
    int e  = blockIdx.y;
    int rt = blockIdx.x;
    int used = cnt[e]; if (used > CAP) used = CAP;
    if (rt * 16 >= used) return;

    __shared__ unsigned short sA[16 * 136];     // input rows (bf16)
    __shared__ unsigned short sBt[128 * 136];   // weight half, transposed [n][k]
    __shared__ unsigned short sH[16 * 264];     // hidden tile 16x256 (bf16)

    int tid = threadIdx.x;
    int lane = tid & 31, wave = tid >> 5;
    int half = lane >> 4, ln = lane & 15;
    const float* W1e = W1 + (size_t)e * DMODEL * DINNER;
    const float* b1e = b1 + (size_t)e * DINNER;
    const float* W2e = W2 + (size_t)e * DINNER * DMODEL;
    const float* b2e = b2 + (size_t)e * DMODEL;

    for (int i = tid; i < 16 * 128; i += 256) {
        int r = i >> 7, c = i & 127;
        int rr = rt * 16 + r;
        sA[r * 136 + c] = (rr < CAP) ? buf[((size_t)e * CAP + rr) * DMODEL + c]
                                     : (unsigned short)0;
    }
    for (int i = tid; i < 128 * 128; i += 256) {       // W1 cols [0,128)
        int k = i >> 7, n = i & 127;
        sBt[n * 136 + k] = f2bf(W1e[(size_t)k * DINNER + n]);
    }
    __syncthreads();

    // GEMM1 + bias + exact GeLU, hidden cols in two 128-wide halves
    for (int hblk = 0; hblk < 2; ++hblk) {
        if (hblk == 1) {
            __syncthreads();
            for (int i = tid; i < 128 * 128; i += 256) {
                int k = i >> 7, n = i & 127;
                sBt[n * 136 + k] = f2bf(W1e[(size_t)k * DINNER + 128 + n]);
            }
            __syncthreads();
        }
        int n0 = wave * 16;
        v8f acc = {};
#pragma unroll
        for (int k0 = 0; k0 < 128; k0 += 32)
            acc = wmma_bf16(frag_a(sA, 136, lane, k0), frag_bt(sBt, 136, lane, k0, n0), acc);
        int col = hblk * 128 + n0 + ln;
#pragma unroll
        for (int r = 0; r < 8; ++r) {
            int m = r + 8 * half;
            float hv = acc[r] + b1e[col];
            float gl = 0.5f * hv * (1.f + erff(hv * 0.70710678118f));
            sH[m * 264 + col] = f2bf(gl);
        }
    }
    __syncthreads();

    // GEMM2: hidden(16x256) @ W2(256x128), K in two 128-deep halves
    v8f acc2 = {};
    int n0 = wave * 16;
    for (int kblk = 0; kblk < 2; ++kblk) {
        for (int i = tid; i < 128 * 128; i += 256) {
            int k = i >> 7, n = i & 127;
            sBt[n * 136 + k] = f2bf(W2e[((size_t)kblk * 128 + k) * DMODEL + n]);
        }
        __syncthreads();
#pragma unroll
        for (int k0 = 0; k0 < 128; k0 += 32)
            acc2 = wmma_bf16(frag_a(sH, 264, lane, kblk * 128 + k0),
                             frag_bt(sBt, 136, lane, k0, n0), acc2);
        __syncthreads();
    }
    int col = n0 + ln;
#pragma unroll
    for (int r = 0; r < 8; ++r) {
        int m = r + 8 * half;
        int rr = rt * 16 + m;
        if (rr < CAP)
            ybuf[((size_t)e * CAP + rr) * DMODEL + col] = f2bf(acc2[r] + b2e[col]);
    }
}

// ---------------- combine + residual ----------------
__global__ __launch_bounds__(128) void k_combine(const float* __restrict__ nf,
                                                 const unsigned short* __restrict__ ybuf,
                                                 const int* __restrict__ slot_map,
                                                 const float* __restrict__ gw,
                                                 float* __restrict__ out) {
    int t = blockIdx.x, j = threadIdx.x;
    float s = nf[(size_t)t * DMODEL + j];
#pragma unroll
    for (int k = 0; k < 2; ++k) {
        int sm = slot_map[t * 2 + k];
        if (sm >= 0) s += gw[t * 2 + k] * bf2f(ybuf[(size_t)sm * DMODEL + j]);
    }
    out[(size_t)t * DMODEL + j] = s;
}

// ---------------- BatchNorm ----------------
__global__ __launch_bounds__(256) void k_bnstats(const float* __restrict__ x,
                                                 float* __restrict__ bn) {
    int c = blockIdx.x;
    __shared__ float ss[256], ss2[256];
    float s = 0.f, s2 = 0.f;
    for (int t = threadIdx.x; t < NNODES; t += 256) {
        float v = x[(size_t)t * DMODEL + c];
        s += v; s2 += v * v;
    }
    ss[threadIdx.x] = s; ss2[threadIdx.x] = s2;
    __syncthreads();
    for (int w = 128; w > 0; w >>= 1) {
        if (threadIdx.x < w) {
            ss[threadIdx.x]  += ss[threadIdx.x + w];
            ss2[threadIdx.x] += ss2[threadIdx.x + w];
        }
        __syncthreads();
    }
    if (threadIdx.x == 0) {
        float mean = ss[0] / (float)NNODES;
        float var  = ss2[0] / (float)NNODES - mean * mean;
        bn[c] = mean; bn[128 + c] = var;
    }
}

__global__ void k_bnapply(float* __restrict__ out, const float* __restrict__ bn,
                          const float* __restrict__ gamma, const float* __restrict__ beta) {
    int n = NNODES * DMODEL;
    for (int i = blockIdx.x * blockDim.x + threadIdx.x; i < n; i += gridDim.x * blockDim.x) {
        int c = i & 127;
        out[i] = (out[i] - bn[c]) * rsqrtf(bn[128 + c] + BNEPS) * gamma[c] + beta[c];
    }
}

extern "C" void kernel_launch(void* const* d_in, const int* in_sizes, int n_in,
                              void* d_out, int out_size, void* d_ws, size_t ws_size,
                              hipStream_t stream) {
    (void)in_sizes; (void)n_in; (void)out_size; (void)ws_size;
    const float* feats = (const float*)d_in[0];
    const int*   esrc  = (const int*)  d_in[1];
    const int*   edst  = (const int*)  d_in[2];
    const float* Wconv = (const float*)d_in[3];
    const float* bconv = (const float*)d_in[4];
    const float* gateW = (const float*)d_in[5];
    const float* gateb = (const float*)d_in[6];
    const float* W1    = (const float*)d_in[7];
    const float* b1    = (const float*)d_in[8];
    const float* W2    = (const float*)d_in[9];
    const float* b2    = (const float*)d_in[10];
    const float* gamma = (const float*)d_in[11];
    const float* beta  = (const float*)d_in[12];
    float* out = (float*)d_out;

    char* ws = (char*)d_ws;
    size_t off = 0;
    auto take = [&](size_t b) { size_t r = off; off += (b + 255) & ~(size_t)255; return r; };
    float* agg           = (float*)(ws + take((size_t)NNODES * DMODEL * 4));
    float* deg           = (float*)(ws + take((size_t)NNODES * 4));
    int*   cnt           = (int*)  (ws + take(NEXP * 4));
    size_t zero_bytes    = off;                       // agg + deg + cnt are contiguous at front
    float* nf            = (float*)(ws + take((size_t)NNODES * DMODEL * 4));
    int*   topi          = (int*)  (ws + take((size_t)PAIRS * 4));
    float* gwbuf         = (float*)(ws + take((size_t)PAIRS * 4));
    int*   slotmap       = (int*)  (ws + take((size_t)PAIRS * 4));
    unsigned short* buf  = (unsigned short*)(ws + take((size_t)NEXP * CAP * DMODEL * 2));
    unsigned short* ybuf = (unsigned short*)(ws + take((size_t)NEXP * CAP * DMODEL * 2));
    float* bn            = (float*)(ws + take(256 * 4));

    hipMemsetAsync(ws, 0, zero_bytes, stream);

    k_degree  <<<512, 256, 0, stream>>>(edst, deg, NEDGES);
    k_scatter <<<NEDGES, 128, 0, stream>>>(feats, esrc, edst, deg, agg);
    k_convgemm<<<NNODES / 16, 256, 0, stream>>>(agg, Wconv, bconv, nf);
    k_gate    <<<NNODES / 4, 256, 0, stream>>>(nf, gateW, gateb, topi, gwbuf);
    k_dispatch<<<PAIRS, 128, 0, stream>>>(nf, topi, buf, slotmap, cnt);
    dim3 eg(NT, NEXP);
    k_expert  <<<eg, 256, 0, stream>>>(buf, W1, b1, W2, b2, cnt, ybuf);
    k_combine <<<NNODES, 128, 0, stream>>>(nf, ybuf, slotmap, gwbuf, out);
    k_bnstats <<<128, 256, 0, stream>>>(out, bn);
    k_bnapply <<<2048, 256, 0, stream>>>(out, bn, gamma, beta);
}
